// QuantLinear_72550587564367
// MI455X (gfx1250) — matrix operand order repaired and checked by reading, not compiled, not run on hardware
//
#include <hip/hip_runtime.h>

typedef _Float16 half_t;
typedef __attribute__((ext_vector_type(2)))  _Float16 h2;
typedef __attribute__((ext_vector_type(8)))  _Float16 v8h;
typedef __attribute__((ext_vector_type(16))) _Float16 v16h;
typedef __attribute__((ext_vector_type(8)))  float    v8f;
typedef __attribute__((ext_vector_type(8)))  unsigned v8u;

#define IN_FEATURES  8192
#define OUT_FEATURES 8192
#define GROUP_SIZE   128
#define N_GROUPS     (IN_FEATURES / GROUP_SIZE)
#define M_TOKENS     16
#define FP16_MAX_F   65504.0f
#define FP8_MAX_F    448.0f

// ---------------------------------------------------------------------------
// Kernel 1: activation quant-dequant, written in nibble-interleaved K order.
// Within each aligned 8-K block, position j holds original K = perm[j],
// perm = {0,4,1,5,2,6,3,7}  (inverse: j -> ((j&3)<<1)|(j>>2)).
// This matches the pair order that falls out of `(w >> 4p) & 0x000F000F`.
// ---------------------------------------------------------------------------
__global__ void qdq_act_kernel(const half_t* __restrict__ x,
                               const half_t* __restrict__ act_scales,
                               half_t* __restrict__ aq, int n) {
    int i = blockIdx.x * blockDim.x + threadIdx.x;
    if (i < n) {
        float s = (float)act_scales[0];
        float v = (float)x[i];
        v = fminf(fmaxf(v, -FP16_MAX_F), FP16_MAX_F);
        v = fminf(fmaxf(v / s, -FP8_MAX_F), FP8_MAX_F) * s;
        int k = i & (IN_FEATURES - 1);
        int m = i >> 13;                       // / IN_FEATURES
        int j = k & 7;
        int dst = (k & ~7) | (((j & 3) << 1) | (j >> 2));
        aq[(size_t)m * IN_FEATURES + dst] = (half_t)v;
    }
}

// one nibble pair, already positioned at bits [3:0]/[19:16] and biased:
// fp16(0x6400 | q) == 1024 + q  ->  (q - z)*s = fma(1024+q, s, -(1024+z)*s)
__device__ __forceinline__ unsigned dq2(unsigned t, h2 s2, h2 nb2) {
    h2 hv = __builtin_bit_cast(h2, t);
    h2 r  = hv * s2 + nb2;                     // v_pk_fma_f16
    return __builtin_bit_cast(unsigned, r);
}

// ---------------------------------------------------------------------------
// Kernel 2: W4 dequant + WMMA GEMM, split-K partials in f32
//   wave = one 16x16 output tile (ntile), blockIdx.y = K-split
//   groups_per_split passed from host (avoids device-side division)
// ---------------------------------------------------------------------------
__global__ void __launch_bounds__(256)
gptq_wmma_kernel(const half_t* __restrict__ aq,        // interleaved
                 const int*    __restrict__ qweight,   // [IN/8, OUT]
                 const int*    __restrict__ qzeros,    // [NG, OUT/8]
                 const half_t* __restrict__ scales,    // [NG, OUT]
                 const int*    __restrict__ g_idx,     // [IN]
                 float*        __restrict__ partial,
                 int groups_per_split) {
    const int lane  = threadIdx.x & 31;
    const int wave  = threadIdx.x >> 5;
    const int ntile = blockIdx.x * 8 + wave;            // 0..511
    const int col   = ntile * 16 + (lane & 15);         // this lane's N column
    const bool hi   = (lane >= 16);

    // A fragment: row M = lane&15; 8-half chunks at K offsets +c and +c+8
    const half_t* arow = aq + (size_t)(lane & 15) * IN_FEATURES;
    const int aoff = hi ? 16 : 0;
    // B fragment: lanes 0-15 cover K k0..k0+15, lanes 16-31 cover k0+16..k0+31
    const int boff = hi ? 16 : 0;

    const unsigned NIB = 0x000F000Fu;   // hoisted to SGPRs by the compiler
    const unsigned BIA = 0x64006400u;

    v8f acc = {};

    const int g0 = blockIdx.y * groups_per_split;

    for (int g = g0; g < g0 + groups_per_split; ++g) {
        // --- per-group metadata (uniform g_idx -> scalar load) ---
        const int gg = g_idx[g * GROUP_SIZE];
        const half_t s = scales[(size_t)gg * OUT_FEATURES + col];
        const int zw = qzeros[gg * (OUT_FEATURES / 8) + (col >> 3)];
        const int z  = ((zw >> ((col & 7) * 4)) & 0xF) + 1;
        const h2 s2  = {s, s};
        const half_t nb = (half_t)(-(float)(1024 + z) * (float)s);
        const h2 nb2 = {nb, nb};

        const int k0 = g * GROUP_SIZE;
        #pragma unroll
        for (int kk = 0; kk < GROUP_SIZE / 32; ++kk) {   // 4 WMMAs / group
            const int kw = k0 + kk * 32;

            // ---- A: two b128 loads of interleaved, post-qdq activations ----
            v8h a_lo = *(const v8h*)(arow + kw + aoff);
            v8h a_hi = *(const v8h*)(arow + kw + aoff + 8);
            v16h a = __builtin_shufflevector(a_lo, a_hi,
                       0,1,2,3,4,5,6,7,8,9,10,11,12,13,14,15);

            // ---- B: 16 packed nibbles = 2 int32 loads ----
            const int r = (kw + boff) >> 3;               // qweight row
            const unsigned w0 = (unsigned)qweight[(size_t)r       * OUT_FEATURES + col];
            const unsigned w1 = (unsigned)qweight[(size_t)(r + 1) * OUT_FEATURES + col];

            // interleaved dequant: pair p of word w = nibbles (p, p+4)
            v8u bw;
            bw[0] = dq2(( w0        & NIB) | BIA, s2, nb2);
            bw[1] = dq2(((w0 >>  4) & NIB) | BIA, s2, nb2);
            bw[2] = dq2(((w0 >>  8) & NIB) | BIA, s2, nb2);
            bw[3] = dq2(((w0 >> 12) & NIB) | BIA, s2, nb2);
            bw[4] = dq2(( w1        & NIB) | BIA, s2, nb2);
            bw[5] = dq2(((w1 >>  4) & NIB) | BIA, s2, nb2);
            bw[6] = dq2(((w1 >>  8) & NIB) | BIA, s2, nb2);
            bw[7] = dq2(((w1 >> 12) & NIB) | BIA, s2, nb2);
            v16h b = __builtin_bit_cast(v16h, bw);

            acc = __builtin_amdgcn_wmma_f32_16x16x32_f16(
                      false, a, false, b, (short)0, acc, false, false);
        }
    }

    // D layout: VGPR r -> M = r (+8 for lanes 16-31), N = lane&15
    float* pout = partial + (size_t)blockIdx.y * (M_TOKENS * OUT_FEATURES)
                          + (size_t)(hi ? 8 : 0) * OUT_FEATURES + col;
    #pragma unroll
    for (int r = 0; r < 8; ++r)
        pout[(size_t)r * OUT_FEATURES] = acc[r];
}

// ---------------------------------------------------------------------------
// Kernel 3: reduce K-splits + bias + fp16-range clamp
// ---------------------------------------------------------------------------
__global__ void epilogue_kernel(const float* __restrict__ partial,
                                const half_t* __restrict__ bias,
                                float* __restrict__ out, int n, int nsplit) {
    int i = blockIdx.x * blockDim.x + threadIdx.x;
    if (i < n) {
        float v = partial[i];
        for (int ksp = 1; ksp < nsplit; ++ksp)
            v += partial[(size_t)ksp * n + i];
        v += (float)bias[i & (OUT_FEATURES - 1)];
        v = fminf(fmaxf(v, -FP16_MAX_F), FP16_MAX_F);
        out[i] = v;
    }
}

// ---------------------------------------------------------------------------
extern "C" void kernel_launch(void* const* d_in, const int* in_sizes, int n_in,
                              void* d_out, int out_size, void* d_ws, size_t ws_size,
                              hipStream_t stream) {
    const half_t* x          = (const half_t*)d_in[0];
    const int*    qweight    = (const int*)d_in[1];
    const int*    qzeros     = (const int*)d_in[2];
    const half_t* scales     = (const half_t*)d_in[3];
    const int*    g_idx      = (const int*)d_in[4];
    const half_t* act_scales = (const half_t*)d_in[5];
    const half_t* bias       = (const half_t*)d_in[6];
    float*        out        = (float*)d_out;

    // workspace: [aq fp16 16x8192 (interleaved) | partial f32 ksplit x 16 x 8192]
    const size_t aq_bytes = (size_t)M_TOKENS * IN_FEATURES * sizeof(half_t);
    half_t* aq      = (half_t*)d_ws;
    float*  partial = (float*)((char*)d_ws + aq_bytes);

    // pick the largest split the workspace allows (occupancy: 512*ksplit waves);
    // partials stay L2-resident (192MB), so split-K traffic is nearly free.
    const size_t slab = (size_t)M_TOKENS * OUT_FEATURES * sizeof(float);
    int ksplit = 8;
    while (ksplit > 1 && aq_bytes + (size_t)ksplit * slab > ws_size) ksplit >>= 1;

    const int n_act = M_TOKENS * IN_FEATURES;       // 131072
    qdq_act_kernel<<<(n_act + 255) / 256, 256, 0, stream>>>(x, act_scales, aq, n_act);

    // 512 N-tiles / 8 waves per block = 64 blocks in x; K-split in y
    dim3 grid(OUT_FEATURES / (16 * 8), ksplit);
    gptq_wmma_kernel<<<grid, 256, 0, stream>>>(aq, qweight, qzeros, scales, g_idx,
                                               partial, N_GROUPS / ksplit);

    const int n_out = M_TOKENS * OUT_FEATURES;      // 131072
    epilogue_kernel<<<(n_out + 255) / 256, 256, 0, stream>>>(partial, bias, out, n_out, ksplit);
}
